// CorrXCA_36077725286475
// MI455X (gfx1250) — matrix operand order repaired and checked by reading, not compiled, not run on hardware
//
#include <hip/hip_runtime.h>

typedef _Float16 v16h __attribute__((ext_vector_type(16)));
typedef float    v8f  __attribute__((ext_vector_type(8)));

#define WDIM 80
#define HDIM 96
#define DDIM 112
#define CCH  16
#define KCH  27
#define MIDC 6
#define ZT   14          // output z-voxels per tile (112 = 8 * 14)
#define NTILE 8
#define WPB  4           // waves per block (wave32)

// x layout: [B,C,W,H,D]  -> ((c*W + x)*H + y)*D + z
// out layout: [B,W,H,D,27] -> (((x*H)+y)*D + z)*27 + k ; k = dx*9+dy*3+dz

__global__ __launch_bounds__(32 * WPB)
void corr_wmma_kernel(const float* __restrict__ x1, const float* __restrict__ x2,
                      float* __restrict__ out, float* __restrict__ sums)
{
    __shared__ float lds[WPB * 9 * 256];
    const int lane = threadIdx.x & 31;
    const int wave = threadIdx.x >> 5;
    const int unit = blockIdx.x * WPB + wave;      // < 80*96*8, exact
    const int zt = unit & 7;
    const int y  = (unit >> 3) % HDIM;
    const int x  = (unit >> 3) / HDIM;
    const int z0 = zt * ZT;
    float* wlds = &lds[wave * 9 * 256];

    // ---- A fragment: rows M = z0+m (m=lane%16), K = channels (0..15), K16..31 zero-padded.
    // f16 A layout: lanes 0-15 hold K=0..7 (vgpr0-3) / K=16..23 (vgpr4-7); lanes 16-31 K=8..15 / 24..31.
    const int m    = lane & 15;
    const int half = lane >> 4;
    const int zA   = z0 + m;
    // rows z >= DDIM only feed discarded G rows -> clamp address, no zeroing needed
    const int zAc  = zA < DDIM ? zA : (DDIM - 1);
    float av[8];
    #pragma unroll
    for (int e = 0; e < 8; ++e)
        av[e] = x1[(((half * 8 + e) * WDIM + x) * HDIM + y) * DDIM + zAc];
    v16h a;
    #pragma unroll
    for (int e = 0; e < 8; ++e) a[e] = (_Float16)av[e];
    #pragma unroll
    for (int e = 8; e < 16; ++e) a[e] = (_Float16)0.0f;

    // ---- 9 Gram products, one per (dx,dy). B columns n=0..15 -> z' = z0-1+n (zero-padded OOB).
    // f16 B (32x16) layout: lanes 0-15 hold column N=lane, K=0..7 / 16..23 (zero); lanes 16-31 K=8..15 / 24..31 (zero).
    const int n   = lane & 15;
    const int zB  = z0 - 1 + n;
    const bool okz = (unsigned)zB < DDIM;
    const int zBc = okz ? zB : 0;
    #pragma unroll
    for (int q = 0; q < 9; ++q) {
        const int dx = q / 3, dy = q % 3;
        const int xx = x + dx - 1;
        const int yy = y + dy - 1;
        const bool okxy = ((unsigned)xx < WDIM) && ((unsigned)yy < HDIM);
        const int xxc = okxy ? xx : x;           // always-valid address
        const int yyc = okxy ? yy : y;
        const float mask = (okxy && okz) ? 1.0f : 0.0f;
        const int pbase = (xxc * HDIM + yyc) * DDIM + zBc;
        float bv[16];
        #pragma unroll
        for (int e = 0; e < 8; ++e)
            bv[e] = x2[(half * 8 + e) * (WDIM * HDIM * DDIM) + pbase];   // unconditional, batched
        #pragma unroll
        for (int e = 8; e < 16; ++e) bv[e] = 0.0f;
        v16h b;
        #pragma unroll
        for (int e = 0; e < 16; ++e) b[e] = (_Float16)(bv[e] * mask);

        v8f acc = {};
        acc = __builtin_amdgcn_wmma_f32_16x16x32_f16(false, a, false, b,
                                                     (short)0, acc, false, false);
        // C/D layout: vgpr v, lanes 0-15 -> (M=v, N=lane); lanes 16-31 -> (M=8+v, N=lane-16)
        #pragma unroll
        for (int v = 0; v < 8; ++v) {
            const int M = v + 8 * half;
            wlds[q * 256 + M * 16 + n] = acc[v];
        }
    }
    __syncthreads();

    // ---- extract near-diagonals: corr[z0+i, k] = G_{k/3}[i, i + k%3] / 16
    const long baseOut = (long)((x * HDIM + y) * DDIM + z0) * KCH;
    for (int t = lane; t < ZT * KCH; t += 32) {
        const int i  = t / KCH;
        const int k  = t - i * KCH;
        const int q  = k / 3;
        const int dz = k - q * 3;
        out[baseOut + t] = wlds[q * 256 + i * 16 + (i + dz)] * 0.0625f;
    }
    // ---- squeeze partial sums (27 channels)
    if (lane < KCH) {
        const int q = lane / 3, dz = lane % 3;
        float s = 0.0f;
        #pragma unroll
        for (int i = 0; i < ZT; ++i) s += wlds[q * 256 + i * 16 + (i + dz)];
        atomicAdd(&sums[lane], s * 0.0625f);
    }
}

__global__ void se_gate_kernel(const float* __restrict__ sums,
                               const float* __restrict__ w0, const float* __restrict__ b0,
                               const float* __restrict__ w1, const float* __restrict__ b1,
                               float* __restrict__ gate)
{
    __shared__ float s[KCH];
    const int t = threadIdx.x;
    if (t < KCH) s[t] = sums[t] * (1.0f / (float)(WDIM * HDIM * DDIM));
    __syncthreads();
    if (t < KCH) {
        float h[MIDC];
        #pragma unroll
        for (int j = 0; j < MIDC; ++j) {
            float acc = b0[j];
            for (int k = 0; k < KCH; ++k) acc += w0[j * KCH + k] * s[k];
            h[j] = acc > 0.0f ? acc : 0.0f;
        }
        float acc = b1[t];
        #pragma unroll
        for (int j = 0; j < MIDC; ++j) acc += w1[t * MIDC + j] * h[j];
        gate[t] = 1.0f / (1.0f + __expf(-acc));
    }
}

__global__ __launch_bounds__(256)
void scale_kernel(float* __restrict__ out, const float* __restrict__ gate, long n4, long n)
{
    __shared__ float g[KCH];
    if (threadIdx.x < KCH) g[threadIdx.x] = gate[threadIdx.x];
    __syncthreads();
    long i = blockIdx.x * (long)blockDim.x + threadIdx.x;
    const long stride = (long)gridDim.x * blockDim.x;
    float4* o4 = (float4*)out;
    for (; i < n4; i += stride) {
        float4 v = o4[i];
        int k = (int)((i * 4) % KCH);
        v.x *= g[k]; k = (k + 1 == KCH) ? 0 : k + 1;
        v.y *= g[k]; k = (k + 1 == KCH) ? 0 : k + 1;
        v.z *= g[k]; k = (k + 1 == KCH) ? 0 : k + 1;
        v.w *= g[k];
        o4[i] = v;
    }
    // scalar tail (out_size divisible by 4 here, but stay safe)
    const long tail = n - n4 * 4;
    if (blockIdx.x == 0 && (long)threadIdx.x < tail) {
        const long idx = n4 * 4 + threadIdx.x;
        out[idx] *= g[(int)(idx % KCH)];
    }
}

extern "C" void kernel_launch(void* const* d_in, const int* in_sizes, int n_in,
                              void* d_out, int out_size, void* d_ws, size_t ws_size,
                              hipStream_t stream)
{
    const float* x1 = (const float*)d_in[0];
    const float* x2 = (const float*)d_in[1];
    const float* w0 = (const float*)d_in[2];
    const float* b0 = (const float*)d_in[3];
    const float* w1 = (const float*)d_in[4];
    const float* b1 = (const float*)d_in[5];
    float* out  = (float*)d_out;
    float* sums = (float*)d_ws;          // 27 accumulators
    float* gate = (float*)d_ws + 64;     // 27 gate values

    hipMemsetAsync(sums, 0, KCH * sizeof(float), stream);

    const int units = WDIM * HDIM * NTILE;   // 61440, divisible by WPB
    corr_wmma_kernel<<<units / WPB, 32 * WPB, 0, stream>>>(x1, x2, out, sums);
    se_gate_kernel<<<1, 32, 0, stream>>>(sums, w0, b0, w1, b1, gate);

    const long n  = (long)out_size;
    const long n4 = n / 4;
    int blocks = (int)((n4 + 255) / 256);
    if (blocks > 16384) blocks = 16384;
    scale_kernel<<<blocks, 256, 0, stream>>>(out, gate, n4, n);
}
